// RelTransformerEncoder_3384434230168
// MI455X (gfx1250) — matrix operand order repaired and checked by reading, not compiled, not run on hardware
//
#include <hip/hip_runtime.h>

// ---------------------------------------------------------------------------
// RelTransformerEncoder for MI455X (gfx1250).
// bf16 WMMA + f32 accumulate; TDM (tensor_load_to_lds) double-buffered tile
// staging with pad-to-LDS-stride; weights pre-transposed at bf16-convert time.
// B=2, I=1024, D=1024, H=16, Dh=64, Krel=64, L=6, F=4096.
// ---------------------------------------------------------------------------

#define BB 2
#define II 1024
#define DD 1024
#define HH 16
#define DH 64
#define LL 6
#define FF 4096
#define KREL 64
#define NBIN 129   // 2K+1

typedef __attribute__((ext_vector_type(16))) __bf16 v16bf;
typedef __attribute__((ext_vector_type(8)))  float  v8f;
typedef __attribute__((ext_vector_type(4)))  int    v4i;

union Frag16 {
    v16bf bf;
    v4i   i4[2];
    unsigned short us[16];
};

__device__ __forceinline__ unsigned short f2bf(float f) {
    union { float f; unsigned u; } x; x.f = f;
    unsigned u = x.u + 0x7FFFu + ((x.u >> 16) & 1u);   // RNE
    return (unsigned short)(u >> 16);
}
__device__ __forceinline__ float bf2f(unsigned short h) {
    union { unsigned u; float f; } x; x.u = ((unsigned)h) << 16;
    return x.f;
}

__device__ __forceinline__ v8f wmma_bf16(const Frag16& a, const Frag16& b, v8f c) {
    return __builtin_amdgcn_wmma_f32_16x16x32_bf16(
        false, a.bf, false, b.bf, (short)0, c, false, false);
}

// ---------------------------------------------------------------------------
// Tensor Data Mover staging (CDNA5).  Guarded: falls back to plain copies.
// This toolchain uses the 6-arg builtin:
//   (uint32x4 g0, int32x8 g1, int32x4 g2, int32x4 g3, int32x8 g4, i32 cpol)
// ---------------------------------------------------------------------------
#if defined(__HIP_DEVICE_COMPILE__) && __has_builtin(__builtin_amdgcn_tensor_load_to_lds) && __has_builtin(__builtin_amdgcn_s_wait_tensorcnt)
#define USE_TDM 1
#else
#define USE_TDM 0
#endif

#if USE_TDM
typedef __attribute__((ext_vector_type(4))) unsigned int v4u;
typedef __attribute__((ext_vector_type(8))) int v8i;

// 2D tile load: tile_d0 elems/row (2B elems), tile_d1 rows, row stride
// stride0 (elems) in global; LDS gets pad_amount DWORDs appended every
// pad_interval DWORDs (codes per D# group1).
__device__ __forceinline__ void tdm_load_2d(const void* gptr, void* lptr,
                                            unsigned tile_d0, unsigned tile_d1,
                                            unsigned stride0, unsigned pic, unsigned pac)
{
    unsigned long long ga = (unsigned long long)gptr;
    unsigned lds = (unsigned)(unsigned long long)lptr;   // low 32 bits = LDS offset
    v4u g0;
    g0.x = 1u;                                            // count=1, user D#
    g0.y = lds;                                           // lds_addr
    g0.z = (unsigned)ga;                                  // global_addr[31:0]
    g0.w = (unsigned)((ga >> 32) & 0x1FFFFFFull) | (2u << 30);  // addr[56:32] | type=2
    unsigned td0 = stride0;        // tensor_dim0 (never OOB: tile starts at addr)
    unsigned td1 = 1u << 20;       // tensor_dim1: large
    v8i g1;
    g1[0] = (int)((1u << 17) | (1u << 20) | (pic << 22) | (pac << 25)); // 2B, pad_en
    g1[1] = (int)((td0 & 0xFFFFu) << 16);                 // tensor_dim0[15:0]
    g1[2] = (int)((td0 >> 16) | ((td1 & 0xFFFFu) << 16)); // dim0[31:16], dim1[15:0]
    g1[3] = (int)((td1 >> 16) | (tile_d0 << 16));         // dim1[31:16], tile_dim0
    g1[4] = (int)tile_d1;                                 // tile_dim1 (tile_dim2=0)
    g1[5] = (int)stride0;                                 // dim0_stride[31:0]
    g1[6] = 0;
    g1[7] = 0;
    v4i zz4 = (v4i){0, 0, 0, 0};
    v8i zz8 = (v8i){0, 0, 0, 0, 0, 0, 0, 0};
    __builtin_amdgcn_tensor_load_to_lds(g0, g1, zz4, zz4, zz8, 0);
}
#endif

// ---------------------------------------------------------------------------
// Embedding + sinusoidal positional encoding -> fp32 x [B*I, D]
// ---------------------------------------------------------------------------
__global__ __launch_bounds__(256) void embed_kernel(
    const int* __restrict__ src, const float* __restrict__ tok,
    float* __restrict__ x)
{
    int row = blockIdx.x;
    int i   = row & (II - 1);
    int t   = src[row];
    const float c = 9.210340371976184f / (float)DD;  // ln(10000)/D
    for (int d = threadIdx.x; d < DD; d += 256) {
        int j = d >> 1;
        float div = __expf(-(float)(2 * j) * c);
        float ang = (float)i * div;
        float pe = (d & 1) ? __cosf(ang) : __sinf(ang);
        x[(size_t)row * DD + d] = tok[(size_t)t * DD + d] * 32.0f + pe;
    }
}

// ---------------------------------------------------------------------------
// fp32 -> bf16 (row-major preserved)  — rel embeddings
// ---------------------------------------------------------------------------
__global__ __launch_bounds__(256) void cvt_kernel(
    const float* __restrict__ in, unsigned short* __restrict__ out, int n)
{
    for (int i = blockIdx.x * 256 + threadIdx.x; i < n; i += gridDim.x * 256)
        out[i] = f2bf(in[i]);
}

// ---------------------------------------------------------------------------
// fp32 [K,N] -> bf16 transposed [N,K]  — weights (LDS-tiled 32x32)
// grid: (N/32, K/32)
// ---------------------------------------------------------------------------
__global__ __launch_bounds__(256) void cvtT_kernel(
    const float* __restrict__ in, unsigned short* __restrict__ out,
    int Kd, int Nd)
{
    __shared__ float tile[32][33];
    int k0 = blockIdx.y * 32, n0 = blockIdx.x * 32;
    int tr = threadIdx.x >> 3;           // 0..31
    int tc = (threadIdx.x & 7) * 4;      // 0,4,..,28
    const float* ip = in + (size_t)(k0 + tr) * Nd + n0 + tc;
    for (int e = 0; e < 4; e++) tile[tr][tc + e] = ip[e];
    __syncthreads();
    unsigned short* op = out + (size_t)(n0 + tr) * Kd + k0 + tc;
    for (int e = 0; e < 4; e++) op[e] = f2bf(tile[tc + e][tr]);
}

// ---------------------------------------------------------------------------
// LayerNorm over D=1024.  One block per row.  Output bf16 or fp32.
// ---------------------------------------------------------------------------
template <bool BF16OUT>
__global__ __launch_bounds__(256) void ln_kernel(
    const float* __restrict__ x, const float* __restrict__ g,
    const float* __restrict__ b, void* __restrict__ out)
{
    int row = blockIdx.x, tid = threadIdx.x;
    const float* xr = x + (size_t)row * DD;
    float v[4], s = 0.f, s2 = 0.f;
    for (int k = 0; k < 4; k++) {
        v[k] = xr[tid + k * 256];
        s += v[k]; s2 += v[k] * v[k];
    }
    __shared__ float r1[256], r2[256];
    r1[tid] = s; r2[tid] = s2;
    __syncthreads();
    for (int st = 128; st > 0; st >>= 1) {
        if (tid < st) { r1[tid] += r1[tid + st]; r2[tid] += r2[tid + st]; }
        __syncthreads();
    }
    float mu = r1[0] * (1.f / DD);
    float var = r2[0] * (1.f / DD) - mu * mu;
    float rs = rsqrtf(var + 1e-5f);
    for (int k = 0; k < 4; k++) {
        int d = tid + k * 256;
        float y = (v[k] - mu) * rs * g[d] + b[d];
        if (BF16OUT) ((unsigned short*)out)[(size_t)row * DD + d] = f2bf(y);
        else         ((float*)out)[(size_t)row * DD + d] = y;
    }
}

// ---------------------------------------------------------------------------
// WMMA GEMM:  C[M,N] = A[M,K](bf16) x Bt[N,K](bf16, pre-transposed) + bias[N]
// 128x128 tile, K-step 32, double-buffered TDM staging.
// RELU: max(0,.) ; RESID: Cf += ; OUTBF: Cb[m,n]=bf16 ; OUTVT: Vt[b][n][i]=bf16
// ---------------------------------------------------------------------------
template <bool RELU, bool RESID, bool OUTBF, bool OUTVT>
__global__ __launch_bounds__(256) void gemm_kernel(
    const unsigned short* __restrict__ A, const unsigned short* __restrict__ Bt,
    const float* __restrict__ bias, float* __restrict__ Cf,
    unsigned short* __restrict__ Cb, int M, int N, int K)
{
    __shared__ __align__(16) unsigned short sA[2][128][40];   // [m][k] pad 32->40
    __shared__ __align__(16) unsigned short sB[2][128][40];   // [n][k] pad 32->40

    const int lane = threadIdx.x & 31, w = threadIdx.x >> 5;
    const int grp = lane >> 4, col = lane & 15;
    const int waveM = w >> 1, waveN = w & 1;
    const int m0 = blockIdx.y * 128, n0 = blockIdx.x * 128;
    const int tRow = threadIdx.x >> 1, tK = (threadIdx.x & 1) * 16;
    (void)tRow; (void)tK;

    v8f acc[2][4];
    for (int i = 0; i < 2; i++)
        for (int j = 0; j < 4; j++)
            acc[i][j] = (v8f){0.f,0.f,0.f,0.f,0.f,0.f,0.f,0.f};

    // ---- stage tile 0 ----
#if USE_TDM
    if (w == 0) {
        tdm_load_2d(A  + (size_t)m0 * K, &sA[0][0][0], 32, 128, (unsigned)K, 3, 3);
        tdm_load_2d(Bt + (size_t)n0 * K, &sB[0][0][0], 32, 128, (unsigned)K, 3, 3);
    }
#else
    {
        const unsigned short* ap = A + ((size_t)(m0 + tRow) * K + tK);
        *(v4i*)(&sA[0][tRow][tK])     = *(const v4i*)(ap);
        *(v4i*)(&sA[0][tRow][tK + 8]) = *(const v4i*)(ap + 8);
        const unsigned short* bp = Bt + ((size_t)(n0 + tRow) * K + tK);
        *(v4i*)(&sB[0][tRow][tK])     = *(const v4i*)(bp);
        *(v4i*)(&sB[0][tRow][tK + 8]) = *(const v4i*)(bp + 8);
    }
#endif

    int buf = 0;
    for (int k0 = 0; k0 < K; k0 += 32, buf ^= 1) {
        const bool more = (k0 + 32) < K;
        if (more) {
#if USE_TDM
            if (w == 0) {
                tdm_load_2d(A  + (size_t)m0 * K + k0 + 32, &sA[buf ^ 1][0][0],
                            32, 128, (unsigned)K, 3, 3);
                tdm_load_2d(Bt + (size_t)n0 * K + k0 + 32, &sB[buf ^ 1][0][0],
                            32, 128, (unsigned)K, 3, 3);
            }
#else
            const unsigned short* ap = A + ((size_t)(m0 + tRow) * K + k0 + 32 + tK);
            *(v4i*)(&sA[buf ^ 1][tRow][tK])     = *(const v4i*)(ap);
            *(v4i*)(&sA[buf ^ 1][tRow][tK + 8]) = *(const v4i*)(ap + 8);
            const unsigned short* bp = Bt + ((size_t)(n0 + tRow) * K + k0 + 32 + tK);
            *(v4i*)(&sB[buf ^ 1][tRow][tK])     = *(const v4i*)(bp);
            *(v4i*)(&sB[buf ^ 1][tRow][tK + 8]) = *(const v4i*)(bp + 8);
            if (k0 + 64 < K) {
                __builtin_prefetch(ap + (size_t)32, 0, 0);
                __builtin_prefetch(bp + (size_t)32, 0, 0);
            }
#endif
        }
#if USE_TDM
        if (w == 0) {
            if (more) __builtin_amdgcn_s_wait_tensorcnt(2);
            else      __builtin_amdgcn_s_wait_tensorcnt(0);
        }
#endif
        __syncthreads();

        Frag16 af[2], bfr[4];
        for (int mi = 0; mi < 2; mi++) {
            int r = waveM * 32 + mi * 16 + col;
            af[mi].i4[0] = *(const v4i*)(&sA[buf][r][grp * 8]);
            af[mi].i4[1] = *(const v4i*)(&sA[buf][r][16 + grp * 8]);
        }
        for (int ni = 0; ni < 4; ni++) {
            int n = waveN * 64 + ni * 16 + col;
            bfr[ni].i4[0] = *(const v4i*)(&sB[buf][n][grp * 8]);
            bfr[ni].i4[1] = *(const v4i*)(&sB[buf][n][16 + grp * 8]);
        }
        for (int mi = 0; mi < 2; mi++)
            for (int ni = 0; ni < 4; ni++)
                acc[mi][ni] = wmma_bf16(af[mi], bfr[ni], acc[mi][ni]);
        __syncthreads();
    }

    // ---- epilogue ----
    for (int mi = 0; mi < 2; mi++)
        for (int ni = 0; ni < 4; ni++) {
            int nn = n0 + waveN * 64 + ni * 16 + col;
            float bval = bias[nn];
            for (int vv = 0; vv < 8; vv++) {
                int mm = m0 + waveM * 32 + mi * 16 + vv + grp * 8;
                float val = acc[mi][ni][vv] + bval;
                if (RELU) val = fmaxf(val, 0.f);
                if (RESID) Cf[(size_t)mm * N + nn] += val;
                if (OUTBF) Cb[(size_t)mm * N + nn] = f2bf(val);
                if (OUTVT) // V^T: [b][n(=dim)][i]
                    Cb[((size_t)(mm >> 10) * DD + nn) * II + (mm & (II - 1))] = f2bf(val);
            }
        }
}

// ---------------------------------------------------------------------------
// Flash attention with relative position bias.
// Grid: (I/128, H, B).  Block 256 = 8 waves; wave w owns 16 query rows.
// q,k: [B*I, D] bf16 (head h at cols h*64);  vT: [B][D][I] bf16.
// ---------------------------------------------------------------------------
__global__ __launch_bounds__(256) void attn_kernel(
    const unsigned short* __restrict__ q, const unsigned short* __restrict__ kmat,
    const unsigned short* __restrict__ vT, const unsigned short* __restrict__ relb,
    unsigned short* __restrict__ out)
{
    const int lane = threadIdx.x & 31, w = threadIdx.x >> 5;
    const int grp = lane >> 4, col = lane & 15;
    const int h = blockIdx.y, b = blockIdx.z;
    const int q0 = blockIdx.x * 128 + w * 16;

    __shared__ __align__(16) unsigned short sArel[128][136]; // bf16 [qrow][bin]
    __shared__ __align__(16) unsigned short sK[2][32][72];   // [key][d] pad 64->72
    __shared__ __align__(16) unsigned short sVt[2][64][40];  // [d][key] pad 32->40
    __shared__ __align__(16) unsigned short sP[8][16][32];   // per-wave P [row][key]

    // ---- Q fragments (16 rows x 64 d -> two K-steps of 32) ----
    Frag16 aq[2];
    {
        const unsigned short* qp = q + ((size_t)(b * II + q0 + col) * DD + h * DH);
        for (int c = 0; c < 2; c++) {
            aq[c].i4[0] = *(const v4i*)(qp + c * 32 + grp * 8);
            aq[c].i4[1] = *(const v4i*)(qp + c * 32 + grp * 8 + 16);
        }
    }

    // ---- rel bias table: a_rel[r, bin] = q[r] . rel[bin] ----
    for (int t = 0; t < 9; t++) {
        int bin = t * 16 + col; if (bin > 128) bin = 128;
        const unsigned short* rp = relb + (size_t)bin * DD + h * DH + grp * 8;
        Frag16 b0, b1;
        b0.i4[0] = *(const v4i*)(rp);       b0.i4[1] = *(const v4i*)(rp + 16);
        b1.i4[0] = *(const v4i*)(rp + 32);  b1.i4[1] = *(const v4i*)(rp + 48);
        v8f acc = (v8f){0.f,0.f,0.f,0.f,0.f,0.f,0.f,0.f};
        acc = wmma_bf16(aq[0], b0, acc);
        acc = wmma_bf16(aq[1], b1, acc);
        int cidx = t * 16 + col;
        if (cidx < 136)
            for (int vv = 0; vv < 8; vv++)
                sArel[w * 16 + vv + grp * 8][cidx] = f2bf(acc[vv]);
        // same-wave LDS write/read: in-order, no barrier needed
    }

    const unsigned short* kbase = kmat + ((size_t)(b * II) * DD + h * DH);
    const unsigned short* vbase = vT + ((size_t)(b * DD + h * DH) * II);

    // ---- stage key-tile 0 ----
#if USE_TDM
    if (w == 0) {
        tdm_load_2d(kbase, &sK[0][0][0],  64, 32, DD, 4, 3);
        tdm_load_2d(vbase, &sVt[0][0][0], 32, 64, II, 3, 3);
    }
#else
    {
        int key = threadIdx.x >> 3, doff = (threadIdx.x & 7) * 8;
        *(v4i*)(&sK[0][key][doff]) = *(const v4i*)(kbase + (size_t)key * DD + doff);
        int d = threadIdx.x >> 2, c = (threadIdx.x & 3) * 8;
        *(v4i*)(&sVt[0][d][c]) = *(const v4i*)(vbase + (size_t)d * II + c);
    }
#endif

    // ---- online softmax state ----
    float m[8], l[8];
    v8f o[4];
    for (int vv = 0; vv < 8; vv++) { m[vv] = -3.0e38f; l[vv] = 0.f; }
    for (int dt = 0; dt < 4; dt++) o[dt] = (v8f){0.f,0.f,0.f,0.f,0.f,0.f,0.f,0.f};
    const float scale = 0.125f;   // 1/sqrt(64)

    int buf = 0;
    for (int j0 = 0; j0 < II; j0 += 32, buf ^= 1) {
        const bool more = (j0 + 32) < II;
        if (more) {
#if USE_TDM
            if (w == 0) {
                tdm_load_2d(kbase + (size_t)(j0 + 32) * DD, &sK[buf ^ 1][0][0],
                            64, 32, DD, 4, 3);
                tdm_load_2d(vbase + (j0 + 32), &sVt[buf ^ 1][0][0],
                            32, 64, II, 3, 3);
            }
#else
            int key = threadIdx.x >> 3, doff = (threadIdx.x & 7) * 8;
            *(v4i*)(&sK[buf ^ 1][key][doff]) =
                *(const v4i*)(kbase + (size_t)(j0 + 32 + key) * DD + doff);
            int d = threadIdx.x >> 2, c = (threadIdx.x & 3) * 8;
            *(v4i*)(&sVt[buf ^ 1][d][c]) =
                *(const v4i*)(vbase + (size_t)d * II + j0 + 32 + c);
#endif
        }
#if USE_TDM
        if (w == 0) {
            if (more) __builtin_amdgcn_s_wait_tensorcnt(2);
            else      __builtin_amdgcn_s_wait_tensorcnt(0);
        }
#endif
        __syncthreads();

        // ---- scores: S = Q x K^T, two 16-key fragments ----
        float sc[2][8];
        for (int f = 0; f < 2; f++) {
            int key = f * 16 + col;
            Frag16 bk0, bk1;
            bk0.i4[0] = *(const v4i*)(&sK[buf][key][grp * 8]);
            bk0.i4[1] = *(const v4i*)(&sK[buf][key][16 + grp * 8]);
            bk1.i4[0] = *(const v4i*)(&sK[buf][key][32 + grp * 8]);
            bk1.i4[1] = *(const v4i*)(&sK[buf][key][48 + grp * 8]);
            v8f s = (v8f){0.f,0.f,0.f,0.f,0.f,0.f,0.f,0.f};
            s = wmma_bf16(aq[0], bk0, s);
            s = wmma_bf16(aq[1], bk1, s);
            for (int vv = 0; vv < 8; vv++) {
                int r = vv + grp * 8;
                int ig = q0 + r;
                int jg = j0 + f * 16 + col;
                int bin = jg - ig;
                bin = bin < -KREL ? -KREL : (bin > KREL ? KREL : bin);
                bin += KREL;
                sc[f][vv] = (s[vv] + bf2f(sArel[w * 16 + r][bin])) * scale;
            }
        }

        // ---- online softmax update (row reductions in half-wave of 16) ----
        float alpha[8];
        for (int vv = 0; vv < 8; vv++) {
            float rm = fmaxf(sc[0][vv], sc[1][vv]);
            for (int off = 1; off < 16; off <<= 1)
                rm = fmaxf(rm, __shfl_xor(rm, off, 16));
            float mn = fmaxf(m[vv], rm);
            float a_ = __expf(m[vv] - mn);
            float p0 = __expf(sc[0][vv] - mn);
            float p1 = __expf(sc[1][vv] - mn);
            sc[0][vv] = p0; sc[1][vv] = p1;
            float rs = p0 + p1;
            for (int off = 1; off < 16; off <<= 1)
                rs += __shfl_xor(rs, off, 16);
            l[vv] = l[vv] * a_ + rs;
            m[vv] = mn;
            alpha[vv] = a_;
        }
        for (int dt = 0; dt < 4; dt++)
            for (int vv = 0; vv < 8; vv++)
                o[dt][vv] *= alpha[vv];

        // ---- P: C-layout -> LDS -> A-layout (16 q-rows x 32 keys) ----
        for (int f = 0; f < 2; f++)
            for (int vv = 0; vv < 8; vv++)
                sP[w][vv + grp * 8][f * 16 + col] = f2bf(sc[f][vv]);
        Frag16 ap;
        ap.i4[0] = *(const v4i*)(&sP[w][col][grp * 8]);
        ap.i4[1] = *(const v4i*)(&sP[w][col][16 + grp * 8]);

        // ---- O += P x V ----
        for (int dt = 0; dt < 4; dt++) {
            int d = dt * 16 + col;
            Frag16 bv;
            bv.i4[0] = *(const v4i*)(&sVt[buf][d][grp * 8]);
            bv.i4[1] = *(const v4i*)(&sVt[buf][d][16 + grp * 8]);
            o[dt] = wmma_bf16(ap, bv, o[dt]);
        }
        __syncthreads();
    }

    // ---- epilogue: normalize by l, write bf16 [B*I, D] head-sliced ----
    for (int vv = 0; vv < 8; vv++) {
        float inv = 1.0f / l[vv];
        int r = vv + grp * 8;
        size_t base = (size_t)(b * II + q0 + r) * DD + h * DH;
        for (int dt = 0; dt < 4; dt++)
            out[base + dt * 16 + col] = f2bf(o[dt][vv] * inv);
    }
}

// ---------------------------------------------------------------------------
// Host orchestration
// ---------------------------------------------------------------------------
extern "C" void kernel_launch(void* const* d_in, const int* in_sizes, int n_in,
                              void* d_out, int out_size, void* d_ws, size_t ws_size,
                              hipStream_t stream)
{
    const int*   src  = (const int*)d_in[0];
    const float* tok  = (const float*)d_in[1];
    const float* rel  = (const float*)d_in[2];
    const float* ln1g = (const float*)d_in[3];
    const float* ln1b = (const float*)d_in[4];
    const float* Wq   = (const float*)d_in[5];
    const float* bq   = (const float*)d_in[6];
    const float* Wk   = (const float*)d_in[7];
    const float* bk   = (const float*)d_in[8];
    const float* Wv   = (const float*)d_in[9];
    const float* bv   = (const float*)d_in[10];
    const float* Wo   = (const float*)d_in[11];
    const float* bo   = (const float*)d_in[12];
    const float* ln2g = (const float*)d_in[13];
    const float* ln2b = (const float*)d_in[14];
    const float* W1   = (const float*)d_in[15];
    const float* b1   = (const float*)d_in[16];
    const float* W2   = (const float*)d_in[17];
    const float* b2   = (const float*)d_in[18];
    const float* lnfg = (const float*)d_in[19];
    const float* lnfb = (const float*)d_in[20];

    const size_t M = (size_t)BB * II;   // 2048 tokens

    // workspace carve-up (all 16B aligned)
    char* p = (char*)d_ws;
    float* x = (float*)p;                      p += M * DD * sizeof(float);
    unsigned short* hbuf = (unsigned short*)p; p += M * DD * 2;
    unsigned short* qb   = (unsigned short*)p; p += M * DD * 2;
    unsigned short* kb   = (unsigned short*)p; p += M * DD * 2;
    unsigned short* vtb  = (unsigned short*)p; p += M * DD * 2;   // V^T [B][D][I]
    unsigned short* aob  = (unsigned short*)p; p += M * DD * 2;
    unsigned short* ffnb = (unsigned short*)p; p += M * FF * 2;
    unsigned short* wqb  = (unsigned short*)p; p += (size_t)DD * DD * 2;  // [N][K]
    unsigned short* wkb  = (unsigned short*)p; p += (size_t)DD * DD * 2;
    unsigned short* wvb  = (unsigned short*)p; p += (size_t)DD * DD * 2;
    unsigned short* wob  = (unsigned short*)p; p += (size_t)DD * DD * 2;
    unsigned short* w1b  = (unsigned short*)p; p += (size_t)DD * FF * 2;  // [F][D]
    unsigned short* w2b  = (unsigned short*)p; p += (size_t)FF * DD * 2;  // [D][F]
    unsigned short* relbuf = (unsigned short*)p; p += (size_t)NBIN * DD * 2;

    dim3 blk(256);
    dim3 gemmDD(DD / 128, (int)(M / 128));
    dim3 gemmDF(FF / 128, (int)(M / 128));
    dim3 attnG(II / 128, HH, BB);
    dim3 ctDD(DD / 32, DD / 32);
    dim3 ctDF(FF / 32, DD / 32);   // W1: in [D,F]
    dim3 ctFD(DD / 32, FF / 32);   // W2: in [F,D]

    embed_kernel<<<(int)M, blk, 0, stream>>>(src, tok, x);

    for (int l = 0; l < LL; l++) {
        const size_t wOff  = (size_t)l * DD * DD;
        const size_t w1Off = (size_t)l * DD * FF;
        const size_t w2Off = (size_t)l * FF * DD;
        const size_t bD    = (size_t)l * DD;
        const size_t bF    = (size_t)l * FF;
        const size_t rOff  = (size_t)l * NBIN * DD;

        // LN1 -> bf16
        ln_kernel<true><<<(int)M, blk, 0, stream>>>(x, ln1g + bD, ln1b + bD, hbuf);

        // weight conversions (transposed to [N][K]) for this layer
        cvtT_kernel<<<ctDD, blk, 0, stream>>>(Wq + wOff, wqb, DD, DD);
        cvtT_kernel<<<ctDD, blk, 0, stream>>>(Wk + wOff, wkb, DD, DD);
        cvtT_kernel<<<ctDD, blk, 0, stream>>>(Wv + wOff, wvb, DD, DD);
        cvtT_kernel<<<ctDD, blk, 0, stream>>>(Wo + wOff, wob, DD, DD);
        cvtT_kernel<<<ctDF, blk, 0, stream>>>(W1 + w1Off, w1b, DD, FF);
        cvtT_kernel<<<ctFD, blk, 0, stream>>>(W2 + w2Off, w2b, FF, DD);
        cvt_kernel<<<516, blk, 0, stream>>>(rel + rOff, relbuf, NBIN * DD);

        // Q, K, V projections (V written transposed per-head for attention)
        gemm_kernel<false, false, true, false><<<gemmDD, blk, 0, stream>>>(
            hbuf, wqb, bq + bD, nullptr, qb, (int)M, DD, DD);
        gemm_kernel<false, false, true, false><<<gemmDD, blk, 0, stream>>>(
            hbuf, wkb, bk + bD, nullptr, kb, (int)M, DD, DD);
        gemm_kernel<false, false, false, true><<<gemmDD, blk, 0, stream>>>(
            hbuf, wvb, bv + bD, nullptr, vtb, (int)M, DD, DD);

        // flash attention with relative bias
        attn_kernel<<<attnG, blk, 0, stream>>>(qb, kb, vtb, relbuf, aob);

        // output projection + residual into x
        gemm_kernel<false, true, false, false><<<gemmDD, blk, 0, stream>>>(
            aob, wob, bo + bD, x, nullptr, (int)M, DD, DD);

        // LN2 -> bf16
        ln_kernel<true><<<(int)M, blk, 0, stream>>>(x, ln2g + bD, ln2b + bD, hbuf);

        // FFN: ReLU(h W1 + b1) W2 + b2, residual into x
        gemm_kernel<true, false, true, false><<<gemmDF, blk, 0, stream>>>(
            hbuf, w1b, b1 + bF, nullptr, ffnb, (int)M, FF, DD);
        gemm_kernel<false, true, false, false><<<gemmDD, blk, 0, stream>>>(
            ffnb, w2b, b2 + bD, x, nullptr, (int)M, DD, FF);
    }

    // final LayerNorm -> fp32 d_out
    ln_kernel<false><<<(int)M, blk, 0, stream>>>(x, lnfg, lnfb, d_out);
}
